// HeteroRGCN_33397665693713
// MI455X (gfx1250) — matrix operand order repaired and checked by reading, not compiled, not run on hardware
//
#include <hip/hip_runtime.h>

typedef __attribute__((ext_vector_type(16))) __bf16 v16bf;
typedef __attribute__((ext_vector_type(2)))  __bf16 v2bf;
typedef __attribute__((ext_vector_type(8)))  float  v8f;
typedef __attribute__((ext_vector_type(8)))  unsigned int v8u;
typedef __attribute__((ext_vector_type(4)))  float  v4f;

// f32 -> bf16 (RNE) packed pair (lo in [15:0], hi in [31:16]).
// Plain casts let the backend pick v_cvt_pk_bf16_f32 if the target has it.
__device__ __forceinline__ unsigned int pk_bf16(float lo, float hi) {
  v2bf r;
  r[0] = (__bf16)lo;
  r[1] = (__bf16)hi;
  return __builtin_bit_cast(unsigned int, r);
}

// Pack W[128,64] (row-major, k-major) into B-fragment-friendly bf16 pairs:
// Wpk[n*64 + p] = pack(W[2p][n], W[2p+1][n])  -> lane's B-frag is 8 contiguous dwords
__global__ void wpack_kernel(const float* __restrict__ W, unsigned int* __restrict__ Wpk) {
  for (int idx = blockIdx.x * blockDim.x + threadIdx.x; idx < 64 * 64;
       idx += gridDim.x * blockDim.x) {
    int n = idx >> 6, p = idx & 63;
    Wpk[idx] = pk_bf16(W[(2 * p) * 64 + n], W[(2 * p + 1) * 64 + n]);
  }
}

__global__ void degree_kernel(const int* __restrict__ dst, float* __restrict__ deg, int E) {
  int e = blockIdx.x * blockDim.x + threadIdx.x;
  if (e < E) atomicAdd(&deg[dst[e]], 1.0f);
}

// C[N,64] = bf16(A[N,128]) @ bf16(W[128,64]) + bias, f32 accumulate via WMMA.
// One wave -> 16 rows x 64 cols (4 accum tiles), K = 4 steps of 32.
// DUAL=1 computes a second output with the same A fragments (follows+clicks share feat_user).
template <int DUAL>
__global__ __launch_bounds__(256) void gemm_bf16_kernel(
    const float* __restrict__ A, int N,
    const unsigned int* __restrict__ Wpk0, const float* __restrict__ bias0, float* __restrict__ C0,
    const unsigned int* __restrict__ Wpk1, const float* __restrict__ bias1, float* __restrict__ C1) {
  const int lane = threadIdx.x & 31;
  const int wave = threadIdx.x >> 5;
  const int row0 = (blockIdx.x * 8 + wave) * 16;
  if (row0 >= N) return;
  const int lo = lane & 15;  // col within tile (B/C), row within tile (A)
  const int hi = lane >> 4;  // K-half selector (A/B), M-half selector (C/D)

  int mload = row0 + lo;
  if (mload > N - 1) mload = N - 1;  // clamp tail-row loads; stores are guarded
  const float* arow = A + (size_t)mload * 128;

  v8f acc0[4] = {};
  v8f acc1[4] = {};

#pragma unroll
  for (int s = 0; s < 4; ++s) {  // K step of 32
    // A fragment: lane holds row (row0+lo); K = s*32 + hi*16 .. +15
    const v4f* ap0 = reinterpret_cast<const v4f*>(arow + s * 32 + hi * 16);
    const v4f* ap1 = reinterpret_cast<const v4f*>(arow + s * 32 + hi * 16 + 8);
    v4f f0 = ap0[0], f1 = ap0[1], f2 = ap1[0], f3 = ap1[1];
    v8u au;
    au[0] = pk_bf16(f0[0], f0[1]);
    au[1] = pk_bf16(f0[2], f0[3]);
    au[2] = pk_bf16(f1[0], f1[1]);
    au[3] = pk_bf16(f1[2], f1[3]);
    au[4] = pk_bf16(f2[0], f2[1]);
    au[5] = pk_bf16(f2[2], f2[3]);
    au[6] = pk_bf16(f3[0], f3[1]);
    au[7] = pk_bf16(f3[2], f3[3]);
    v16bf afrag = __builtin_bit_cast(v16bf, au);

#pragma unroll
    for (int t = 0; t < 4; ++t) {  // N tile of 16
      const int boff = (t * 16 + lo) * 64 + s * 16 + hi * 8;  // 32B aligned
      v16bf bfrag0 = __builtin_bit_cast(v16bf, *reinterpret_cast<const v8u*>(Wpk0 + boff));
      acc0[t] = __builtin_amdgcn_wmma_f32_16x16x32_bf16(
          false, afrag, false, bfrag0, (short)0, acc0[t], false, false);
      if (DUAL) {
        v16bf bfrag1 = __builtin_bit_cast(v16bf, *reinterpret_cast<const v8u*>(Wpk1 + boff));
        acc1[t] = __builtin_amdgcn_wmma_f32_16x16x32_bf16(
            false, afrag, false, bfrag1, (short)0, acc1[t], false, false);
      }
    }
  }

// C/D layout: lanes 0-15 vgpr i -> M=i, N=lo ; lanes 16-31 vgpr i -> M=8+i, N=lo
#pragma unroll
  for (int t = 0; t < 4; ++t) {
    float bv0 = bias0[t * 16 + lo];
    float bv1 = DUAL ? bias1[t * 16 + lo] : 0.0f;
#pragma unroll
    for (int i = 0; i < 8; ++i) {
      int r = row0 + hi * 8 + i;
      if (r < N) {
        C0[(size_t)r * 64 + t * 16 + lo] = acc0[t][i] + bv0;
        if (DUAL) C1[(size_t)r * 64 + t * 16 + lo] = acc1[t][i] + bv1;
      }
    }
  }
}

// per-(edge,col) scatter of deg-normalized contributions (mean fused via 1/deg[dst])
__global__ void scatter64_kernel(const float* __restrict__ Wh, const int* __restrict__ src,
                                 const int* __restrict__ dst, const float* __restrict__ deg,
                                 float* __restrict__ acc, int E) {
  long gid = (long)blockIdx.x * blockDim.x + threadIdx.x;
  if (gid >= (long)E * 64) return;
  int e = (int)(gid >> 6);
  int c = (int)(gid & 63);
  int s = src[e], d = dst[e];
  float dg = deg[d];
  float inv = (dg > 0.0f) ? (1.0f / dg) : 0.0f;
  atomicAdd(&acc[(size_t)d * 64 + c], Wh[(size_t)s * 64 + c] * inv);
}

__global__ void scatter2_kernel(const float* __restrict__ Wh, const int* __restrict__ src,
                                const int* __restrict__ dst, const float* __restrict__ deg,
                                float* __restrict__ acc, int E) {
  int e = blockIdx.x * blockDim.x + threadIdx.x;
  if (e >= E) return;
  int s = src[e], d = dst[e];
  float dg = deg[d];
  float inv = (dg > 0.0f) ? (1.0f / dg) : 0.0f;
  atomicAdd(&acc[(size_t)d * 2 + 0], Wh[(size_t)s * 2 + 0] * inv);
  atomicAdd(&acc[(size_t)d * 2 + 1], Wh[(size_t)s * 2 + 1] * inv);
}

__global__ void leaky_kernel(float* __restrict__ x, long n) {
  long i = (long)blockIdx.x * blockDim.x + threadIdx.x;
  if (i < n) {
    float v = x[i];
    x[i] = (v >= 0.0f) ? v : 0.01f * v;
  }
}

// out[n, 0:2] = h[n, 0:64] @ W2[64,2] + b2
__global__ void linear2_kernel(const float* __restrict__ Hh, const float* __restrict__ W2,
                               const float* __restrict__ b2, float* __restrict__ out, int N) {
  int n = blockIdx.x * blockDim.x + threadIdx.x;
  if (n >= N) return;
  const float* h = Hh + (size_t)n * 64;
  float a0 = 0.0f, a1 = 0.0f;
#pragma unroll
  for (int k = 0; k < 64; ++k) {
    float hv = h[k];
    a0 += hv * W2[k * 2 + 0];
    a1 += hv * W2[k * 2 + 1];
  }
  out[(size_t)n * 2 + 0] = a0 + b2[0];
  out[(size_t)n * 2 + 1] = a1 + b2[1];
}

extern "C" void kernel_launch(void* const* d_in, const int* in_sizes, int n_in,
                              void* d_out, int out_size, void* d_ws, size_t ws_size,
                              hipStream_t stream) {
  (void)n_in; (void)ws_size;
  const float* fu  = (const float*)d_in[0];
  const float* fi  = (const float*)d_in[1];
  const float* W1f = (const float*)d_in[2];  const float* b1f = (const float*)d_in[3];
  const float* W1c = (const float*)d_in[4];  const float* b1c = (const float*)d_in[5];
  const float* W1r = (const float*)d_in[6];  const float* b1r = (const float*)d_in[7];
  const float* W2f = (const float*)d_in[8];  const float* b2f = (const float*)d_in[9];
  const float* W2c = (const float*)d_in[10]; const float* b2c = (const float*)d_in[11];
  const float* W2r = (const float*)d_in[12]; const float* b2r = (const float*)d_in[13];
  const int* fsrc = (const int*)d_in[14]; const int* fdst = (const int*)d_in[15];
  const int* csrc = (const int*)d_in[16]; const int* cdst = (const int*)d_in[17];
  const int* rsrc = (const int*)d_in[18]; const int* rdst = (const int*)d_in[19];

  const int Nu = in_sizes[0] / 128;
  const int Ni = in_sizes[1] / 128;
  const int E  = in_sizes[14];

  // d_out layout: out_u[Nu*2] | out_i[Ni*2] | h2u[Nu*64] | h2i[Ni*64]
  float* out   = (float*)d_out;
  float* out_u = out;
  float* out_i = out + (size_t)2 * Nu;
  float* h2u   = out + (size_t)2 * Nu + (size_t)2 * Ni;
  float* h2i   = h2u + (size_t)64 * Nu;

  // workspace (~201 MB): packed weights | degrees | Wh1 | Wh2
  float* wsf = (float*)d_ws;
  unsigned int* wpk_f = (unsigned int*)wsf;
  unsigned int* wpk_c = wpk_f + 4096;
  unsigned int* wpk_r = wpk_c + 4096;
  float* deg_f = wsf + 12288;          // [Nu]
  float* deg_c = deg_f + Nu;           // [Ni]
  float* deg_r = deg_c + Ni;           // [Nu]
  float* Wh1_f = deg_r + Nu;           // [Nu*64]
  float* Wh1_c = Wh1_f + (size_t)64 * Nu;
  float* Wh1_r = Wh1_c + (size_t)64 * Nu;  // [Ni*64]
  float* Wh2_f = Wh1_r + (size_t)64 * Ni;  // [Nu*2]
  float* Wh2_c = Wh2_f + (size_t)2 * Nu;
  float* Wh2_r = Wh2_c + (size_t)2 * Nu;   // [Ni*2]

  hipMemsetAsync(d_out, 0, (size_t)out_size * sizeof(float), stream);
  hipMemsetAsync(deg_f, 0, ((size_t)2 * Nu + (size_t)Ni) * sizeof(float), stream);

  wpack_kernel<<<16, 256, 0, stream>>>(W1f, wpk_f);
  wpack_kernel<<<16, 256, 0, stream>>>(W1c, wpk_c);
  wpack_kernel<<<16, 256, 0, stream>>>(W1r, wpk_r);

  const int eb = (E + 255) / 256;
  degree_kernel<<<eb, 256, 0, stream>>>(fdst, deg_f, E);
  degree_kernel<<<eb, 256, 0, stream>>>(cdst, deg_c, E);
  degree_kernel<<<eb, 256, 0, stream>>>(rdst, deg_r, E);

  // Layer-1 GEMMs (bf16 WMMA, f32 accumulate)
  gemm_bf16_kernel<1><<<(Nu + 127) / 128, 256, 0, stream>>>(fu, Nu, wpk_f, b1f, Wh1_f,
                                                            wpk_c, b1c, Wh1_c);
  gemm_bf16_kernel<0><<<(Ni + 127) / 128, 256, 0, stream>>>(fi, Ni, wpk_r, b1r, Wh1_r,
                                                            nullptr, nullptr, nullptr);

  // Layer-1 mean-aggregation: follows + rev share the user accumulator (pre-normalized)
  const long st = (long)E * 64;
  const int sb = (int)((st + 255) / 256);
  scatter64_kernel<<<sb, 256, 0, stream>>>(Wh1_f, fsrc, fdst, deg_f, h2u, E);
  scatter64_kernel<<<sb, 256, 0, stream>>>(Wh1_r, rsrc, rdst, deg_r, h2u, E);
  scatter64_kernel<<<sb, 256, 0, stream>>>(Wh1_c, csrc, cdst, deg_c, h2i, E);

  // LeakyReLU in place over h2u|h2i (contiguous)
  const long hl = (long)64 * ((long)Nu + Ni);
  leaky_kernel<<<(int)((hl + 255) / 256), 256, 0, stream>>>(h2u, hl);

  // Layer-2 linears (K=64, C=2 -> plain VALU)
  linear2_kernel<<<(Nu + 255) / 256, 256, 0, stream>>>(h2u, W2f, b2f, Wh2_f, Nu);
  linear2_kernel<<<(Nu + 255) / 256, 256, 0, stream>>>(h2u, W2c, b2c, Wh2_c, Nu);
  linear2_kernel<<<(Ni + 255) / 256, 256, 0, stream>>>(h2i, W2r, b2r, Wh2_r, Ni);

  // Layer-2 mean-aggregation into final outputs
  scatter2_kernel<<<eb, 256, 0, stream>>>(Wh2_f, fsrc, fdst, deg_f, out_u, E);
  scatter2_kernel<<<eb, 256, 0, stream>>>(Wh2_r, rsrc, rdst, deg_r, out_u, E);
  scatter2_kernel<<<eb, 256, 0, stream>>>(Wh2_c, csrc, cdst, deg_c, out_i, E);
}
